// SO2_Linear_55319178772734
// MI455X (gfx1250) — matrix operand order repaired and checked by reading, compiled, not run on hardware
//
#include <hip/hip_runtime.h>
#include <cmath>

typedef __attribute__((ext_vector_type(16))) __bf16 v16bf;
typedef __attribute__((ext_vector_type(8)))  float  v8f;
typedef unsigned int u32x4 __attribute__((ext_vector_type(4)));
typedef int          i32x8 __attribute__((ext_vector_type(8)));
typedef int          i32x4 __attribute__((ext_vector_type(4)));

#define NSAMP 16384
#define DIMX  3200
#define ROTSZ 164

// ---------------------------------------------------------------- bf16 helpers
__device__ __forceinline__ unsigned short f2bf(float f) {
    unsigned int u = __float_as_uint(f);
    unsigned int r = u + 0x7FFFu + ((u >> 16) & 1u);   // round-to-nearest-even
    return (unsigned short)(r >> 16);
}

// ---------------------------------------------------------------- J matrices
struct Cx { float re, im; };
__device__ __forceinline__ Cx cxmul(Cx a, Cx b) {
    Cx r; r.re = a.re*b.re - a.im*b.im; r.im = a.re*b.im + a.im*b.re; return r;
}

// One thread per l (l = 1..4): builds SO(3) generators via SU(2) + real<->complex
// basis change, then J_l = expm(pi*(X0+X1)/sqrt(2)) with order-24 series + 10 squarings.
__global__ void compute_J_kernel(float* __restrict__ Jbuf) {
    const int l = (int)threadIdx.x + 1;
    if (l > 4) return;
    const int joff[5] = {0, 0, 9, 34, 83};
    const int d = 2*l + 1;

    float rais[81], low[81];
    for (int i = 0; i < 81; ++i) { rais[i] = 0.f; low[i] = 0.f; }
    const float jf = (float)l;
    for (int i = 0; i < d-1; ++i) {
        float m = -jf + (float)i;
        rais[(i+1)*d + i] = -sqrtf(jf*(jf+1.f) - m*(m+1.f));
    }
    for (int i = 0; i < d-1; ++i) {
        float m = -jf + 1.f + (float)i;
        low[i*d + (i+1)] = sqrtf(jf*(jf+1.f) - m*(m-1.f));
    }
    Cx X[81];
    for (int i = 0; i < d; ++i)
        for (int j = 0; j < d; ++j) {
            X[i*d+j].re = 0.5f*(rais[i*d+j] + low[i*d+j]);
            X[i*d+j].im = (i == j) ? (float)(-l + i) : 0.f;
        }
    Cx q[81];
    for (int i = 0; i < 81; ++i) { q[i].re = 0.f; q[i].im = 0.f; }
    const float s2 = 0.70710678118654752f;
    for (int m = -l; m < 0; ++m) {
        q[(l+m)*d + (l-m)].re = s2;
        q[(l+m)*d + (l+m)].im = -s2;
    }
    q[l*d + l].re = 1.f;
    for (int m = 1; m <= l; ++m) {
        float sg = (m & 1) ? -1.f : 1.f;
        q[(l+m)*d + (l+m)].re = sg*s2;
        q[(l+m)*d + (l-m)].im = sg*s2;
    }
    Cx ph;
    switch (l & 3) {                       // (-i)^l
        case 0: ph.re =  1.f; ph.im =  0.f; break;
        case 1: ph.re =  0.f; ph.im = -1.f; break;
        case 2: ph.re = -1.f; ph.im =  0.f; break;
        default:ph.re =  0.f; ph.im =  1.f; break;
    }
    for (int i = 0; i < d*d; ++i) q[i] = cxmul(ph, q[i]);

    Cx T[81];
    for (int i = 0; i < d; ++i)
        for (int j = 0; j < d; ++j) {
            Cx acc; acc.re = 0.f; acc.im = 0.f;
            for (int k = 0; k < d; ++k) {
                Cx p = cxmul(X[i*d+k], q[k*d+j]);
                acc.re += p.re; acc.im += p.im;
            }
            T[i*d+j] = acc;
        }
    float A[81];
    const float scale = 3.14159265358979323846f * 0.70710678118654752f / 1024.f;
    for (int i = 0; i < d; ++i)
        for (int j = 0; j < d; ++j) {
            float acc = 0.f;
            for (int k = 0; k < d; ++k)
                acc += q[k*d+i].re*T[k*d+j].re + q[k*d+i].im*T[k*d+j].im;
            A[i*d+j] = acc * scale;
        }
    float E[81], P[81], tmp[81];
    for (int i = 0; i < d; ++i)
        for (int j = 0; j < d; ++j) { float v = (i==j)?1.f:0.f; E[i*d+j]=v; P[i*d+j]=v; }
    for (int k = 1; k < 24; ++k) {
        float inv = 1.f/(float)k;
        for (int i = 0; i < d; ++i)
            for (int j = 0; j < d; ++j) {
                float acc = 0.f;
                for (int t = 0; t < d; ++t) acc += P[i*d+t]*A[t*d+j];
                tmp[i*d+j] = acc*inv;
            }
        for (int e = 0; e < d*d; ++e) { P[e] = tmp[e]; E[e] += tmp[e]; }
    }
    for (int it = 0; it < 10; ++it) {
        for (int i = 0; i < d; ++i)
            for (int j = 0; j < d; ++j) {
                float acc = 0.f;
                for (int t = 0; t < d; ++t) acc += E[i*d+t]*E[t*d+j];
                tmp[i*d+j] = acc;
            }
        for (int e = 0; e < d*d; ++e) E[e] = tmp[e];
    }
    float* dst = Jbuf + joff[l];
    for (int e = 0; e < d*d; ++e) dst[e] = E[e];
}

// ---------------------------------------------------------------- per-sample Wigner D
__global__ void wigner_kernel(const float* __restrict__ R,
                              const float* __restrict__ Jbuf,
                              float* __restrict__ rotBuf) {
    const int n = blockIdx.x*blockDim.x + threadIdx.x;
    if (n >= NSAMP) return;
    float r0 = R[n*3+0], r1 = R[n*3+1], r2 = R[n*3+2];
    float vx = r1, vy = r2, vz = r0;
    float nn = sqrtf(vx*vx + vy*vy + vz*vz);
    nn = fmaxf(nn, 1e-12f);
    vx /= nn; vy /= nn; vz /= nn;
    vx = fminf(fmaxf(vx, -1.f), 1.f);
    vy = fminf(fmaxf(vy, -1.f), 1.f);
    vz = fminf(fmaxf(vz, -1.f), 1.f);
    const float beta  = acosf(vy);
    const float alpha = atan2f(vx, vz);
    const int joff[5] = {0, 0, 9, 34, 83};
    float* out = rotBuf + (size_t)n*ROTSZ;
    for (int l = 1; l <= 4; ++l) {
        const int d = 2*l + 1;
        const float* J = Jbuf + joff[l];
        float cb[9], sb[9], ca[9], sa[9];
        for (int a = 0; a < d; ++a) {
            float f = (float)(l - a);
            cb[a] = cosf(f*beta);  sb[a] = sinf(f*beta);
            ca[a] = cosf(f*alpha); sa[a] = sinf(f*alpha);
        }
        float M1[81], M2[81];
        for (int i = 0; i < d; ++i)
            for (int j = 0; j < d; ++j)
                M1[i*d+j] = J[i*d+j]*cb[j] + J[i*d + (d-1-j)]*sb[d-1-j];
        for (int i = 0; i < d; ++i)
            for (int j = 0; j < d; ++j) {
                float acc = 0.f;
                for (int t = 0; t < d; ++t) acc += M1[i*d+t]*J[t*d+j];
                M2[i*d+j] = acc;
            }
        float* Dl = out + joff[l];
        for (int i = 0; i < d; ++i)
            for (int j = 0; j < d; ++j)
                Dl[i*d+j] = ca[i]*M2[i*d+j] + sa[i]*M2[(d-1-i)*d + j];
    }
}

// ---------------------------------------------------------------- weight prep (bf16)
__global__ void convert_w0_kernel(const float* __restrict__ w,
                                  unsigned short* __restrict__ dst, int n) {
    int i = blockIdx.x*blockDim.x + threadIdx.x;
    if (i < n) dst[i] = f2bf(w[i]);
}

// M = [[WA, -WB],[WB, WA]], WA = w[0:co,:], WB = w[co:2co,:]
__global__ void build_m_kernel(const float* __restrict__ w, int co,
                               unsigned short* __restrict__ dst) {
    const int K = 2*co;
    const int i = blockIdx.x*blockDim.x + threadIdx.x;
    if (i >= K*K) return;
    const int o = i / K, t = i - o*K;
    float v;
    if (t < co)          v =  w[o*co + t];
    else if (o < co)     v = -w[(o+co)*co + (t-co)];
    else                 v =  w[(o-co)*co + (t-co)];
    dst[i] = f2bf(v);
}

// ---------------------------------------------------------------- rotate forward + gather
// Gathered bf16 row layout (width 3200): [F0:640 | G1:1024 | G2:768 | G3:512 | G4:256]
__global__ __launch_bounds__(256) void rotate_gather_kernel(
        const float* __restrict__ x, const float* __restrict__ rotBuf,
        unsigned short* __restrict__ Gbuf) {
    __shared__ float xs[DIMX];
    __shared__ float rs[ROTSZ];
    const int n = blockIdx.x, tid = threadIdx.x;
    const float* xrow = x + (size_t)n*DIMX;
    for (int i = tid; i < DIMX; i += 256) xs[i] = xrow[i];
    const float* rrow = rotBuf + (size_t)n*ROTSZ;
    for (int i = tid; i < ROTSZ; i += 256) rs[i] = rrow[i];
    __syncthreads();
    unsigned short* grow = Gbuf + (size_t)n*DIMX;
    for (int i = tid; i < 128; i += 256) grow[i] = 0;   // l=0 part of x_rot is zeroed
    const int offL[5] = {0, 128, 512, 1152, 2048};
    const int joff[5] = {0, 0, 9, 34, 83};
    const int gOff[5] = {0, 640, 1664, 2432, 2944};
    for (int l = 1; l <= 4; ++l) {
        const int d = 2*l + 1;
        const float* Dm = rs + joff[l];
        const float* xb = xs + offL[l];
        const int tot = 128*d;
        for (int idx = tid; idx < tot; idx += 256) {
            const int k = idx / d, i = idx - k*d;
            float acc = 0.f;
            for (int j = 0; j < d; ++j) acc += Dm[j*d + i]*xb[k*d + j];   // D^T x
            const int mm = i - l;
            if (mm == 0) grow[l*128 + k] = f2bf(acc);
            else {
                const int m = mm < 0 ? -mm : mm;
                grow[gOff[m] + (l-m)*256 + k*2 + (mm > 0 ? 1 : 0)] = f2bf(acc);
            }
        }
    }
}

// ---------------------------------------------------------------- bf16 WMMA GEMM
// Out[n][o] = sum_t G[n][t]*W[o][t] (+bias[o]); tile 128x64 per block, K-chunks of 32.
// A tile: TDM tensor_load_to_lds (TENSORcnt, wave0 issues one descriptor per tile).
// W tile: per-lane global_load_async_to_lds_b128 (ASYNCcnt). Double-buffered.
__global__ __launch_bounds__(256) void gemm_bf16_kernel(
        const unsigned short* __restrict__ G, int gcol,
        const unsigned short* __restrict__ W, int Kdim,
        float* __restrict__ Out, int ocol,
        const float* __restrict__ bias) {
    __shared__ __align__(16) unsigned short Alds[2][128*32];
    __shared__ __align__(16) unsigned short Wlds[2][64*32];
    const int tid  = threadIdx.x;
    const int lane = tid & 31;
    const int wv   = tid >> 5;
    const int mBase = blockIdx.y * 128;
    const int nBase = blockIdx.x * 64;
    const int nK = Kdim >> 5;

    v8f acc0 = {}, acc1 = {}, acc2 = {}, acc3 = {};

    const int wRowL = tid >> 2, wQ = tid & 3;      // W copier: 4 thr/row, 8 cols (16B) each
    const unsigned short* wSrc = W + (size_t)(nBase + wRowL)*Kdim + wQ*8;
    unsigned wDst0 = (unsigned)(size_t)&Wlds[0][wRowL*32 + wQ*8];
    unsigned wDst1 = (unsigned)(size_t)&Wlds[1][wRowL*32 + wQ*8];

    const int fragRow = lane & 15;
    const int fragHi  = lane >> 4;

    // per-lane async b128 for the W tile of K-chunk kb into buffer b
    auto issue_w = [&](int kb, int b) {
        const unsigned short* gw = wSrc + kb*32;
        unsigned wd = b ? wDst1 : wDst0;
        asm volatile("global_load_async_to_lds_b128 %0, %1, off"
                     :: "v"(wd), "v"(gw) : "memory");
    };

    // TDM descriptor: 2D tile 32(K, contiguous) x 128(rows, stride DIMX) bf16 -> Alds[b]
    auto issue_a_tdm = [&](int kb, int b) {
        unsigned lds = (unsigned)(size_t)&Alds[b][0];
        unsigned long long ga = (unsigned long long)(size_t)G
                              + 2ull*((size_t)mBase*DIMX + (size_t)gcol + (size_t)kb*32);
        u32x4 g0;
        g0[0] = 1u;                                         // count=1, user mode
        g0[1] = lds;                                        // lds_addr (bytes)
        g0[2] = (unsigned)ga;                               // global_addr[31:0]
        g0[3] = ((unsigned)(ga >> 32) & 0x01FFFFFFu) | 0x80000000u; // [56:32] | type=2
        const unsigned td0 = 32u, td1 = 128u;               // tensor dims == tile dims
        const unsigned tile0 = 32u, tile1 = 128u;
        const unsigned long long str0 = (unsigned long long)DIMX;   // dim0 stride (elems)
        i32x8 g1;
        g1[0] = 0x00010000;                                 // wg_mask=0, data_size=1 (2B)
        g1[1] = (int)((td0 & 0xFFFFu) << 16);               // tensor_dim0[15:0] @ bits63:48
        g1[2] = (int)(((td0 >> 16) & 0xFFFFu) | ((td1 & 0xFFFFu) << 16));
        g1[3] = (int)(((td1 >> 16) & 0xFFFFu) | ((tile0 & 0xFFFFu) << 16));
        g1[4] = (int)(tile1 & 0xFFFFu);                     // tile_dim1; tile_dim2=0
        g1[5] = (int)(unsigned)(str0 & 0xFFFFFFFFu);        // dim0_stride[31:0]
        g1[6] = (int)(unsigned)((str0 >> 32) & 0xFFFFu);    // dim0_stride[47:32]; dim1_stride lo=0
        g1[7] = 0;
        i32x4 g2 = {0, 0, 0, 0};
        i32x4 g3 = {0, 0, 0, 0};
        i32x8 g4 = {0, 0, 0, 0, 0, 0, 0, 0};                // extra group (clang-23 6-arg form)
        __builtin_amdgcn_tensor_load_to_lds(g0, g1, g2, g3, g4, 0);
    };

    if (wv == 0) issue_a_tdm(0, 0);
    issue_w(0, 0);
    for (int kb = 0; kb < nK; ++kb) {
        const int buf = kb & 1;
        if (kb + 1 < nK) {
            if (wv == 0) issue_a_tdm(kb + 1, buf ^ 1);
            issue_w(kb + 1, buf ^ 1);
            // in-order completion per counter: <=1 outstanding => current tile landed
            asm volatile("s_wait_asynccnt 0x1" ::: "memory");
            if (wv == 0) __builtin_amdgcn_s_wait_tensorcnt((short)1);
        } else {
            asm volatile("s_wait_asynccnt 0x0" ::: "memory");
            if (wv == 0) __builtin_amdgcn_s_wait_tensorcnt((short)0);
        }
        __syncthreads();

        const unsigned short* Ab = &Alds[buf][0];
        const unsigned short* Wb = &Wlds[buf][0];
        // A fragment (16x32 bf16): lane<16 -> K {0..7,16..23}, lane>=16 -> K {8..15,24..31}
        union { uint4 q[2]; v16bf v; } af;
        const int ar = wv*16 + fragRow;
        af.q[0] = *(const uint4*)(Ab + ar*32 + fragHi*8);
        af.q[1] = *(const uint4*)(Ab + ar*32 + 16 + fragHi*8);
        #pragma unroll
        for (int nsub = 0; nsub < 4; ++nsub) {
            // B fragment (32x16): col = lane&15, lane<16 -> K {0..7,16..23}, lane>=16 -> K {8..15,24..31}
            union { uint4 q[2]; v16bf v; } bfrag;
            const int br = nsub*16 + fragRow;
            bfrag.q[0] = *(const uint4*)(Wb + br*32 + fragHi*8);
            bfrag.q[1] = *(const uint4*)(Wb + br*32 + 16 + fragHi*8);
            v8f* ap = (nsub == 0) ? &acc0 : (nsub == 1) ? &acc1 : (nsub == 2) ? &acc2 : &acc3;
            *ap = __builtin_amdgcn_wmma_f32_16x16x32_bf16(
                      false, af.v, false, bfrag.v, (short)0, *ap, false, false);
        }
        __syncthreads();   // everyone done reading buf before it is refilled
    }
    // epilogue: C layout = lane<16: N=lane, M=e; lane>=16: N=lane-16, M=8+e
    const int colLocal = lane & 15;
    const int rowBase  = mBase + wv*16 + fragHi*8;
    #pragma unroll
    for (int nsub = 0; nsub < 4; ++nsub) {
        const v8f a = (nsub == 0) ? acc0 : (nsub == 1) ? acc1 : (nsub == 2) ? acc2 : acc3;
        const int col = nBase + nsub*16 + colLocal;
        const float bv = bias ? bias[col] : 0.f;
        float* o = Out + (size_t)rowBase*DIMX + ocol + col;
        #pragma unroll
        for (int e = 0; e < 8; ++e) o[(size_t)e*DIMX] = a[e] + bv;
    }
}

// ---------------------------------------------------------------- rotate back (in-place row)
__global__ __launch_bounds__(256) void rotate_back_kernel(
        const float* __restrict__ rotBuf, float* __restrict__ out) {
    __shared__ float os[DIMX];
    __shared__ float rs[ROTSZ];
    const int n = blockIdx.x, tid = threadIdx.x;
    float* orow = out + (size_t)n*DIMX;
    for (int i = tid; i < DIMX; i += 256) os[i] = orow[i];   // gathered layout from GEMMs
    const float* rrow = rotBuf + (size_t)n*ROTSZ;
    for (int i = tid; i < ROTSZ; i += 256) rs[i] = rrow[i];
    __syncthreads();
    for (int i = tid; i < 128; i += 256) orow[i] = os[i];    // l=0 passthrough
    const int offL[5] = {0, 128, 512, 1152, 2048};
    const int joff[5] = {0, 0, 9, 34, 83};
    const int gOff[5] = {0, 640, 1664, 2432, 2944};
    for (int l = 1; l <= 4; ++l) {
        const int d = 2*l + 1;
        const float* Dm = rs + joff[l];
        const int tot = 128*d;
        for (int idx = tid; idx < tot; idx += 256) {
            const int k = idx / d, i = idx - k*d;
            float acc = 0.f;
            for (int j = 0; j < d; ++j) {
                const int mm = j - l;
                float v;
                if (mm == 0) v = os[l*128 + k];
                else {
                    const int m = mm < 0 ? -mm : mm;
                    v = os[gOff[m] + (l-m)*256 + k*2 + (mm > 0 ? 1 : 0)];
                }
                acc += Dm[j*d + i]*v;   // D^T applied again
            }
            orow[offL[l] + k*d + i] = acc;
        }
    }
}

// ---------------------------------------------------------------- launch
extern "C" void kernel_launch(void* const* d_in, const int* in_sizes, int n_in,
                              void* d_out, int out_size, void* d_ws, size_t ws_size,
                              hipStream_t stream) {
    const float* x     = (const float*)d_in[0];
    const float* R     = (const float*)d_in[1];
    const float* fc0_w = (const float*)d_in[2];
    const float* fc0_b = (const float*)d_in[3];
    const float* w1    = (const float*)d_in[4];
    const float* w2    = (const float*)d_in[5];
    const float* w3    = (const float*)d_in[6];
    const float* w4    = (const float*)d_in[7];
    float* out = (float*)d_out;
    (void)in_sizes; (void)n_in; (void)out_size; (void)ws_size;

    char* ws = (char*)d_ws;
    size_t off = 0;
    auto alloc = [&](size_t bytes) {
        size_t p = off; off += (bytes + 255) & ~(size_t)255; return p;
    };
    float* Jbuf   = (float*)(ws + alloc(ROTSZ*sizeof(float)));
    float* rotBuf = (float*)(ws + alloc((size_t)NSAMP*ROTSZ*sizeof(float)));
    unsigned short* W0 = (unsigned short*)(ws + alloc((size_t)640*640*2));
    unsigned short* M1 = (unsigned short*)(ws + alloc((size_t)1024*1024*2));
    unsigned short* M2 = (unsigned short*)(ws + alloc((size_t)768*768*2));
    unsigned short* M3 = (unsigned short*)(ws + alloc((size_t)512*512*2));
    unsigned short* M4 = (unsigned short*)(ws + alloc((size_t)256*256*2));
    unsigned short* Gbuf = (unsigned short*)(ws + alloc((size_t)NSAMP*DIMX*2));

    compute_J_kernel<<<1, 4, 0, stream>>>(Jbuf);
    wigner_kernel<<<(NSAMP + 127)/128, 128, 0, stream>>>(R, Jbuf, rotBuf);
    convert_w0_kernel<<<(640*640 + 255)/256, 256, 0, stream>>>(fc0_w, W0, 640*640);
    build_m_kernel<<<(1024*1024 + 255)/256, 256, 0, stream>>>(w1, 512, M1);
    build_m_kernel<<<(768*768  + 255)/256, 256, 0, stream>>>(w2, 384, M2);
    build_m_kernel<<<(512*512  + 255)/256, 256, 0, stream>>>(w3, 256, M3);
    build_m_kernel<<<(256*256  + 255)/256, 256, 0, stream>>>(w4, 128, M4);
    rotate_gather_kernel<<<NSAMP, 256, 0, stream>>>(x, rotBuf, Gbuf);

    gemm_bf16_kernel<<<dim3(640/64,  NSAMP/128), 256, 0, stream>>>(Gbuf, 0,    W0, 640,  out, 0,    fc0_b);
    gemm_bf16_kernel<<<dim3(1024/64, NSAMP/128), 256, 0, stream>>>(Gbuf, 640,  M1, 1024, out, 640,  nullptr);
    gemm_bf16_kernel<<<dim3(768/64,  NSAMP/128), 256, 0, stream>>>(Gbuf, 1664, M2, 768,  out, 1664, nullptr);
    gemm_bf16_kernel<<<dim3(512/64,  NSAMP/128), 256, 0, stream>>>(Gbuf, 2432, M3, 512,  out, 2432, nullptr);
    gemm_bf16_kernel<<<dim3(256/64,  NSAMP/128), 256, 0, stream>>>(Gbuf, 2944, M4, 256,  out, 2944, nullptr);

    rotate_back_kernel<<<NSAMP, 256, 0, stream>>>(rotBuf, out);
}
